// vivi_mamba_87600152969236
// MI455X (gfx1250) — matrix operand order repaired and checked by reading, not compiled, not run on hardware
//
#include <hip/hip_runtime.h>
#include <math.h>

// ---- problem dims (fixed by reference) ----
#define DEPTH  4
#define DM     384
#define DI     768
#define DSTATE 16
#define DTR    24
#define DCONV  4
#define BB     4
#define LL     1024
#define ML     (BB*LL)   // 4096 tokens

typedef __bf16 bf16;
typedef __attribute__((ext_vector_type(16))) __bf16 v16bf;
typedef __attribute__((ext_vector_type(8)))  float  v8f;

// =====================================================================
// CDNA5 async global->LDS copy (ASYNCcnt-tracked, no VGPR data)
// =====================================================================
__device__ __forceinline__ void async_ld_b128(unsigned lds_off, unsigned long long gaddr) {
  asm volatile("global_load_async_to_lds_b128 %0, %1, off"
               :: "v"(lds_off), "v"(gaddr) : "memory");
}
__device__ __forceinline__ void wait_async_le3() {
  asm volatile("s_wait_asynccnt 0x3" ::: "memory");
}

// =====================================================================
// Elementwise / conversion kernels
// =====================================================================
__global__ void k_cvt_bf16(const float* __restrict__ in, bf16* __restrict__ out, int n) {
  int i = blockIdx.x * blockDim.x + threadIdx.x;
  if (i < n) out[i] = (bf16)in[i];
}

// x_proj_w (l,56,DI) -> padded bf16 (l,64,DI), rows 56..63 zero
__global__ void k_pad_xproj(const float* __restrict__ w, bf16* __restrict__ out) {
  int i = blockIdx.x * blockDim.x + threadIdx.x;
  const int n = DEPTH * 64 * DI;
  if (i >= n) return;
  int c = i % DI;
  int r = (i / DI) % 64;
  int l = i / (64 * DI);
  const int R = DTR + 2 * DSTATE; // 56
  out[i] = (bf16)((r < R) ? w[((size_t)l * R + r) * DI + c] : 0.f);
}

// dt_proj_w (l,DI,24) -> padded bf16 (l,DI,32), cols 24..31 zero
__global__ void k_pad_dtw(const float* __restrict__ w, bf16* __restrict__ out) {
  int i = blockIdx.x * blockDim.x + threadIdx.x;
  const int n = DEPTH * DI * 32;
  if (i >= n) return;
  int c = i & 31;
  int r = (i >> 5) % DI;
  int l = i / (DI * 32);
  out[i] = (bf16)((c < DTR) ? w[((size_t)l * DI + r) * DTR + c] : 0.f);
}

// dbl (ML x 64) cols 0..23 -> dt_bf (ML x 32) padded
__global__ void k_dtprep(const float* __restrict__ dbl, bf16* __restrict__ dtb, int n) {
  int i = blockIdx.x * blockDim.x + threadIdx.x;
  if (i >= n) return;
  int c = i & 31;
  int row = i >> 5;
  dtb[i] = (bf16)((c < DTR) ? dbl[(size_t)row * 64 + c] : 0.f);
}

__global__ void k_softplus_bias(float* __restrict__ buf, const float* __restrict__ bias, int n) {
  int i = blockIdx.x * blockDim.x + threadIdx.x;
  if (i >= n) return;
  int d = i % DI;
  float x = buf[i] + bias[d];
  buf[i] = (x > 20.f) ? x : log1pf(__expf(x));
}

// =====================================================================
// Fused residual-add + RMSNorm (one 384-wide row per block, 128 thr)
// =====================================================================
__global__ __launch_bounds__(128) void k_add_rmsnorm(
    const float* __restrict__ hin, float* __restrict__ residual,
    const float* __restrict__ w, bf16* __restrict__ out_bf,
    float* __restrict__ out_f, int addRes, int storeRes) {
  const int row = blockIdx.x;
  const int t = threadIdx.x;
  float r[3];
  float ss = 0.f;
#pragma unroll
  for (int j = 0; j < 3; ++j) {
    int c = t + j * 128;
    size_t idx = (size_t)row * DM + c;
    float v = hin[idx];
    if (addRes) v += residual[idx];
    if (storeRes) residual[idx] = v;
    r[j] = v;
    ss += v * v;
  }
#pragma unroll
  for (int m = 16; m >= 1; m >>= 1) ss += __shfl_xor(ss, m, 32);
  __shared__ float red[4];
  if ((t & 31) == 0) red[t >> 5] = ss;
  __syncthreads();
  float tot = red[0] + red[1] + red[2] + red[3];
  float scale = rsqrtf(tot / (float)DM + 1e-5f);
#pragma unroll
  for (int j = 0; j < 3; ++j) {
    int c = t + j * 128;
    size_t idx = (size_t)row * DM + c;
    float o = r[j] * scale * w[c];
    if (out_bf) out_bf[idx] = (bf16)o;
    if (out_f)  out_f[idx]  = o;
  }
}

// =====================================================================
// bf16 WMMA GEMM:  C(MxN, f32) = A(MxK, bf16 row-major) * W(NxK, bf16)^T
// block = 8 waves, 128x64 tile, wave = 32x32 (2x2 WMMA 16x16x32), K step 32
// Tiles double-buffered in LDS, filled with async global->LDS b128 copies.
// Branch-free inner loop: prefetch is always issued (K-offset clamped on the
// last iteration -> redundant dead fill), single s_wait_asynccnt 3.
// =====================================================================
#define BM 128
#define BN 64
#define BK 32
#define ABYTES (BM*BK*2)   // 8192
#define WBYTES (BN*BK*2)   // 4096

__global__ __launch_bounds__(256) void k_gemm_bf16(
    const bf16* __restrict__ A, const bf16* __restrict__ W,
    float* __restrict__ C, int M, int N, int K) {
  __shared__ __align__(16) bf16 sA[2][BM * BK]; // 2 x 8 KB
  __shared__ __align__(16) bf16 sW[2][BN * BK]; // 2 x 4 KB
  const int tid  = threadIdx.x;
  const int lane = tid & 31;
  const int wave = tid >> 5;
  const int half = lane >> 4;   // lanes 16-31 -> 1
  const int l15  = lane & 15;
  const int bm = blockIdx.y * BM;
  const int bn = blockIdx.x * BN;
  const int wm = (wave & 3) * 32;
  const int wn = (wave >> 2) * 32;

  const v8f vzero = {0.f, 0.f, 0.f, 0.f, 0.f, 0.f, 0.f, 0.f};
  v8f acc[2][2];
  acc[0][0] = vzero; acc[0][1] = vzero; acc[1][0] = vzero; acc[1][1] = vzero;

  // per-thread async-copy slots: 2 x 16B of A tile, 1 x 16B of W tile
  const int u0 = tid, u1 = tid + 256;
  const bf16* gA0 = A + (size_t)(bm + (u0 >> 2)) * K + (u0 & 3) * 8;
  const bf16* gA1 = A + (size_t)(bm + (u1 >> 2)) * K + (u1 & 3) * 8;
  const bf16* gW  = W + (size_t)(bn + (tid >> 2)) * K + (tid & 3) * 8;
  const unsigned sAbase = (unsigned)(size_t)(&sA[0][0]);
  const unsigned sWbase = (unsigned)(size_t)(&sW[0][0]);
  const unsigned lA0 = sAbase + (unsigned)u0 * 16;
  const unsigned lA1 = sAbase + (unsigned)u1 * 16;
  const unsigned lW  = sWbase + (unsigned)tid * 16;

  // prologue: fill buffer 0 with tile 0
  async_ld_b128(lA0, (unsigned long long)(size_t)gA0);
  async_ld_b128(lA1, (unsigned long long)(size_t)gA1);
  async_ld_b128(lW,  (unsigned long long)(size_t)gW);

  const int nk = K / BK;
  for (int kb = 0; kb < nk; ++kb) {
    const int p = kb & 1;
    // always prefetch (clamped on last iteration -> dead fill, never read)
    const int koff = (kb + 1 < nk) ? (kb + 1) * BK : kb * BK;
    async_ld_b128(lA0 + (p ^ 1) * ABYTES, (unsigned long long)(size_t)(gA0 + koff));
    async_ld_b128(lA1 + (p ^ 1) * ABYTES, (unsigned long long)(size_t)(gA1 + koff));
    async_ld_b128(lW  + (p ^ 1) * WBYTES, (unsigned long long)(size_t)(gW  + koff));
    wait_async_le3();   // own wave: tile kb done (async loads complete in order)
    __syncthreads();    // other waves' fills visible

    // fragment assembly per ISA 7.12.2:
    // A 16x32 bf16: lane(l15=M, half): chunk0 K=8*half..+7, chunk1 K=16+8*half..+7
    // B 32x16 bf16: lane(l15=N, half): chunk0 K=8*half..+7, chunk1 K=16+8*half..+7
    const uint4* aq = (const uint4*)sA[p];
    const uint4* wq = (const uint4*)sW[p];
    union Frag { uint4 q[2]; v16bf v; };
    Frag af[2], bfr[2];
#pragma unroll
    for (int mi = 0; mi < 2; ++mi) {
      const uint4* rp = aq + (wm + mi * 16 + l15) * 4;
      af[mi].q[0] = rp[half];
      af[mi].q[1] = rp[2 + half];
    }
#pragma unroll
    for (int ni = 0; ni < 2; ++ni) {
      const uint4* rp = wq + (wn + ni * 16 + l15) * 4;
      bfr[ni].q[0] = rp[half];
      bfr[ni].q[1] = rp[2 + half];
    }
#pragma unroll
    for (int mi = 0; mi < 2; ++mi)
#pragma unroll
      for (int ni = 0; ni < 2; ++ni)
        acc[mi][ni] = __builtin_amdgcn_wmma_f32_16x16x32_bf16(
            false, af[mi].v, false, bfr[ni].v, (short)0, acc[mi][ni], false, false);
    __syncthreads();      // protect buffer before next fill
  }
  // 3 dangling (dead) asyncs are flushed by S_ENDPGM's implicit wait-idle.

  // C layout: VGPR i -> M = i + 8*half, N = lane&15
#pragma unroll
  for (int mi = 0; mi < 2; ++mi)
#pragma unroll
    for (int ni = 0; ni < 2; ++ni) {
      int col  = bn + wn + ni * 16 + l15;
      int rowb = bm + wm + mi * 16 + 8 * half;
#pragma unroll
      for (int i = 0; i < 8; ++i)
        C[(size_t)(rowb + i) * N + col] = acc[mi][ni][i];
    }
}

// =====================================================================
// Causal depthwise conv (K=4) + SiLU, both directions in one pass.
// Backward outputs are stored in SCAN order (reversed sequence).
// =====================================================================
__global__ void k_conv_silu(
    const float* __restrict__ xz,
    const float* __restrict__ cwf, const float* __restrict__ cbf,
    const float* __restrict__ cwb, const float* __restrict__ cbb,
    float* __restrict__ xcf, float* __restrict__ xcb,
    bf16* __restrict__ xcf_bf, bf16* __restrict__ xcb_bf) {
  int i = blockIdx.x * blockDim.x + threadIdx.x;
  if (i >= ML * DI) return;
  int d = i % DI;
  int t = (i / DI) % LL;
  int b = i / (DI * LL);
  const size_t base = (size_t)b * LL;

  float accf = cbf[d];
#pragma unroll
  for (int j = 0; j < DCONV; ++j) {
    int tt = t - (DCONV - 1) + j;
    if (tt >= 0) accf += cwf[d * DCONV + j] * xz[(base + tt) * (2 * DI) + d];
  }
  float sf = accf / (1.f + __expf(-accf));

  float accb = cbb[d];
#pragma unroll
  for (int j = 0; j < DCONV; ++j) {
    int tt = (LL - 1 - t) + (DCONV - 1) - j; // reversed-sequence causal tap
    if (tt < LL) accb += cwb[d * DCONV + j] * xz[(base + tt) * (2 * DI) + d];
  }
  float sb = accb / (1.f + __expf(-accb));

  xcf[i] = sf; xcf_bf[i] = (bf16)sf;
  xcb[i] = sb; xcb_bf[i] = (bf16)sb;
}

// =====================================================================
// Selective scan: lane = (d-sub, n).  h_t = exp(dt*A)*h + dt*xc*B;
// 16-lane shuffle reduction for y = sum_n h*C, gated by silu(z).
// forward: writes y_sum (f32).
// backward: inputs in scan (reversed) order; reads forward y_sum at the
// re-reversed position, adds, and stores the bf16 out_proj input directly.
// =====================================================================
__global__ __launch_bounds__(256) void k_scan(
    const float* __restrict__ delta, const float* __restrict__ xc,
    const float* __restrict__ dbl, const float* __restrict__ xz,
    const float* __restrict__ A_log, const float* __restrict__ Dw,
    float* __restrict__ y_sum, bf16* __restrict__ y_bf, int backward) {
  const int n    = threadIdx.x & 15;
  const int dl   = threadIdx.x >> 4;
  const int dgrp = blockIdx.x % (DI / 16);
  const int b    = blockIdx.x / (DI / 16);
  const int d    = dgrp * 16 + dl;
  const float Aval = -__expf(A_log[d * DSTATE + n]);
  const float Dval = Dw[d];
  float h = 0.f;
  const size_t rowbase = (size_t)b * LL;
  for (int t = 0; t < LL; ++t) {
    size_t ridx = rowbase + t;
    float dt = delta[ridx * DI + d];
    float xv = xc[ridx * DI + d];
    float Bn = dbl[ridx * 64 + DTR + n];
    float Cn = dbl[ridx * 64 + DTR + DSTATE + n];
    float dA = __expf(dt * Aval);
    h = dA * h + (dt * xv) * Bn;
    float p = h * Cn;
    p += __shfl_xor(p, 1, 32);
    p += __shfl_xor(p, 2, 32);
    p += __shfl_xor(p, 4, 32);
    p += __shfl_xor(p, 8, 32);
    if (n == 0) {
      int to = backward ? (LL - 1 - t) : t;
      size_t oidx = rowbase + to;
      float z = xz[oidx * (2 * DI) + DI + d];
      float gate = z / (1.f + __expf(-z));
      float y = (p + xv * Dval) * gate;
      if (backward) y_bf[oidx * DI + d] = (bf16)(y_sum[oidx * DI + d] + y);
      else          y_sum[oidx * DI + d] = y;
    }
  }
}

// =====================================================================
// Host orchestration
// =====================================================================
extern "C" void kernel_launch(void* const* d_in, const int* in_sizes, int n_in,
                              void* d_out, int out_size, void* d_ws, size_t ws_size,
                              hipStream_t stream) {
  (void)in_sizes; (void)n_in; (void)out_size; (void)ws_size;
  const float* x          = (const float*)d_in[0];
  const float* norm_w     = (const float*)d_in[1];
  const float* in_proj_w  = (const float*)d_in[2];
  const float* conv_w     = (const float*)d_in[3];
  const float* conv_b     = (const float*)d_in[4];
  const float* x_proj_w   = (const float*)d_in[5];
  const float* dt_proj_w  = (const float*)d_in[6];
  const float* dt_proj_b  = (const float*)d_in[7];
  const float* A_log      = (const float*)d_in[8];
  const float* Dw         = (const float*)d_in[9];
  const float* conv_w_b   = (const float*)d_in[10];
  const float* conv_b_b   = (const float*)d_in[11];
  const float* x_proj_w_b = (const float*)d_in[12];
  const float* dt_proj_w_b= (const float*)d_in[13];
  const float* dt_proj_b_b= (const float*)d_in[14];
  const float* A_b_log    = (const float*)d_in[15];
  const float* D_b        = (const float*)d_in[16];
  const float* out_proj_w = (const float*)d_in[17];
  const float* norm_f_w   = (const float*)d_in[18];

  char* ws = (char*)d_ws;
  size_t off = 0;
  auto alloc = [&](size_t bytes) -> char* {
    char* p = ws + off;
    off += (bytes + 255) & ~(size_t)255;
    return p;
  };
  float* residual = (float*)alloc((size_t)ML * DM * 4);
  float* hidden   = (float*)alloc((size_t)ML * DM * 4);
  bf16*  h_bf     = (bf16*) alloc((size_t)ML * DM * 2);
  float* xzbuf    = (float*)alloc((size_t)ML * 2 * DI * 4);
  float* xcf      = (float*)alloc((size_t)ML * DI * 4);
  float* xcb      = (float*)alloc((size_t)ML * DI * 4);
  bf16*  xcf_bf   = (bf16*) alloc((size_t)ML * DI * 2);
  bf16*  xcb_bf   = (bf16*) alloc((size_t)ML * DI * 2);
  float* dbl_f    = (float*)alloc((size_t)ML * 64 * 4);
  float* dbl_b    = (float*)alloc((size_t)ML * 64 * 4);
  bf16*  dtbf_f   = (bf16*) alloc((size_t)ML * 32 * 2);
  bf16*  dtbf_b   = (bf16*) alloc((size_t)ML * 32 * 2);
  float* delta_f  = (float*)alloc((size_t)ML * DI * 4);
  float* delta_b  = (float*)alloc((size_t)ML * DI * 4);
  float* y_sum    = (float*)alloc((size_t)ML * DI * 4);
  bf16*  y_bf     = (bf16*) alloc((size_t)ML * DI * 2);
  bf16*  w_in_bf  = (bf16*) alloc((size_t)DEPTH * 2 * DI * DM * 2);
  bf16*  w_out_bf = (bf16*) alloc((size_t)DEPTH * DM * DI * 2);
  bf16*  w_xpf_bf = (bf16*) alloc((size_t)DEPTH * 64 * DI * 2);
  bf16*  w_xpb_bf = (bf16*) alloc((size_t)DEPTH * 64 * DI * 2);
  bf16*  w_dtf_bf = (bf16*) alloc((size_t)DEPTH * DI * 32 * 2);
  bf16*  w_dtb_bf = (bf16*) alloc((size_t)DEPTH * DI * 32 * 2);

  auto cdiv = [](int a, int b) { return (a + b - 1) / b; };

  // one-time (per call) weight down-conversion to bf16 (+ K/N padding)
  { int n = DEPTH * 2 * DI * DM; k_cvt_bf16<<<cdiv(n,256),256,0,stream>>>(in_proj_w,  w_in_bf,  n); }
  { int n = DEPTH * DM * DI;     k_cvt_bf16<<<cdiv(n,256),256,0,stream>>>(out_proj_w, w_out_bf, n); }
  { int n = DEPTH * 64 * DI;
    k_pad_xproj<<<cdiv(n,256),256,0,stream>>>(x_proj_w,   w_xpf_bf);
    k_pad_xproj<<<cdiv(n,256),256,0,stream>>>(x_proj_w_b, w_xpb_bf); }
  { int n = DEPTH * DI * 32;
    k_pad_dtw<<<cdiv(n,256),256,0,stream>>>(dt_proj_w,   w_dtf_bf);
    k_pad_dtw<<<cdiv(n,256),256,0,stream>>>(dt_proj_w_b, w_dtb_bf); }

  for (int l = 0; l < DEPTH; ++l) {
    const float* hin = (l == 0) ? x : hidden;
    // residual += hidden ; h = rmsnorm(residual) -> bf16
    k_add_rmsnorm<<<ML,128,0,stream>>>(hin, residual, norm_w + (size_t)l * DM,
                                       h_bf, nullptr, (l > 0) ? 1 : 0, 1);
    // xz = h @ in_proj_w^T   (4096 x 1536 x 384)
    k_gemm_bf16<<<dim3((2*DI)/BN, ML/BM),256,0,stream>>>(
        h_bf, w_in_bf + (size_t)l * 2 * DI * DM, xzbuf, ML, 2 * DI, DM);
    // depthwise causal conv + silu, both directions
    { int n = ML * DI;
      k_conv_silu<<<cdiv(n,256),256,0,stream>>>(xzbuf,
          conv_w   + (size_t)l * DI * DCONV, conv_b   + (size_t)l * DI,
          conv_w_b + (size_t)l * DI * DCONV, conv_b_b + (size_t)l * DI,
          xcf, xcb, xcf_bf, xcb_bf); }
    // x_proj: (4096 x 64 x 768) per direction
    k_gemm_bf16<<<dim3(64/BN, ML/BM),256,0,stream>>>(
        xcf_bf, w_xpf_bf + (size_t)l * 64 * DI, dbl_f, ML, 64, DI);
    k_gemm_bf16<<<dim3(64/BN, ML/BM),256,0,stream>>>(
        xcb_bf, w_xpb_bf + (size_t)l * 64 * DI, dbl_b, ML, 64, DI);
    // dt part -> bf16 (K padded to 32), dt_proj GEMM, softplus(+bias)
    { int n = ML * 32;
      k_dtprep<<<cdiv(n,256),256,0,stream>>>(dbl_f, dtbf_f, n);
      k_dtprep<<<cdiv(n,256),256,0,stream>>>(dbl_b, dtbf_b, n); }
    k_gemm_bf16<<<dim3(DI/BN, ML/BM),256,0,stream>>>(
        dtbf_f, w_dtf_bf + (size_t)l * DI * 32, delta_f, ML, DI, 32);
    k_gemm_bf16<<<dim3(DI/BN, ML/BM),256,0,stream>>>(
        dtbf_b, w_dtb_bf + (size_t)l * DI * 32, delta_b, ML, DI, 32);
    { int n = ML * DI;
      k_softplus_bias<<<cdiv(n,256),256,0,stream>>>(delta_f, dt_proj_b   + (size_t)l * DI, n);
      k_softplus_bias<<<cdiv(n,256),256,0,stream>>>(delta_b, dt_proj_b_b + (size_t)l * DI, n); }
    // selective scans (fwd writes f32, bwd adds + emits bf16 out_proj input)
    k_scan<<<BB*(DI/16),256,0,stream>>>(delta_f, xcf, dbl_f, xzbuf,
        A_log   + (size_t)l * DI * DSTATE, Dw  + (size_t)l * DI, y_sum, y_bf, 0);
    k_scan<<<BB*(DI/16),256,0,stream>>>(delta_b, xcb, dbl_b, xzbuf,
        A_b_log + (size_t)l * DI * DSTATE, D_b + (size_t)l * DI, y_sum, y_bf, 1);
    // out_proj: (4096 x 384 x 768)
    k_gemm_bf16<<<dim3(DM/BN, ML/BM),256,0,stream>>>(
        y_bf, w_out_bf + (size_t)l * DM * DI, hidden, ML, DM, DI);
  }
  // final: out = rmsnorm(hidden + residual) * norm_f_w  (fp32)
  k_add_rmsnorm<<<ML,128,0,stream>>>(hidden, residual, norm_f_w,
                                     nullptr, (float*)d_out, 1, 0);
}
